// GatheringLoss_26001732010460
// MI455X (gfx1250) — matrix operand order repaired and checked by reading, not compiled, not run on hardware
//
#include <hip/hip_runtime.h>
#include <hip/hip_bf16.h>

// Problem dims (fixed by the reference)
#define D_MODEL    512
#define M_ITEMS    4096
#define ROWS_TOT   32768                        // B*N = 32*1024

// Pipeline: one stage = 2 WMMA column tiles = 32 item rows = 32 KB bf16
#define STAGE_ROWS  32
#define STAGE_BYTES (STAGE_ROWS * D_MODEL * 2)  // 32768
#define TILE_BYTES  (16 * D_MODEL * 2)          // 16384 (one 16-col tile)
#define N_STAGES    (M_ITEMS / STAGE_ROWS)      // 128

typedef __attribute__((ext_vector_type(16))) __bf16 v16bf;
typedef __attribute__((ext_vector_type(8)))  float  v8f;

// fp32 -> bf16, round-to-nearest-even
static __device__ __forceinline__ unsigned short bfbits(float f) {
  unsigned u = __float_as_uint(f);
  u = (u + 0x7FFFu + ((u >> 16) & 1u)) >> 16;
  return (unsigned short)u;
}
static __device__ __forceinline__ __bf16 f2bf(float f) {
  unsigned short s = bfbits(f);
  __bf16 r;
  __builtin_memcpy(&r, &s, 2);
  return r;
}

// ---------------------------------------------------------------------------
// Kernel 1: convert items fp32 -> bf16 (4 elements / thread)
// ---------------------------------------------------------------------------
__global__ void cvt_items_kernel(const float* __restrict__ in,
                                 unsigned short* __restrict__ out) {
  int i = blockIdx.x * blockDim.x + threadIdx.x;
  float4 v = ((const float4*)in)[i];
  ushort4 o;
  o.x = bfbits(v.x); o.y = bfbits(v.y); o.z = bfbits(v.z); o.w = bfbits(v.w);
  ((ushort4*)out)[i] = o;
}

// ---------------------------------------------------------------------------
// Kernel 2: zero the f64 accumulator
// ---------------------------------------------------------------------------
__global__ void init_kernel(double* __restrict__ accum) { *accum = 0.0; }

// ---------------------------------------------------------------------------
// Kernel 3: fused WMMA GEMM + argmax with async-to-LDS double buffering.
//
// Block = 256 threads = 8 waves; each wave owns 16 query rows (A pinned in
// 128 VGPRs as bf16 fragments). The block cooperatively stages each 32-row
// item stage into LDS *in WMMA B-fragment order* via
// GLOBAL_LOAD_ASYNC_TO_LDS_B128 (ASYNCcnt), double-buffered, so every B tile
// is read from L2 once per block instead of once per wave (8x less traffic).
// Two independent accumulator chains per stage keep the XDL pipe busy.
//
// LDS layout per buffer: chunk index c = tile*512 + kc*32 + lane, 32 B each:
//   holds items[stage*32 + tile*16 + (lane&15)],
//         bf16 cols [kc*32 + (lane>>4)*16 .. +16)
// which is each lane's 32 B B fragment in contiguous k = 16*(lane>>4) + s order, as A uses.
// ---------------------------------------------------------------------------
__global__ __launch_bounds__(256, 1)
void gemm_argmax_kernel(const float* __restrict__ q,
                        const unsigned short* __restrict__ itemsBf,
                        int* __restrict__ idxOut) {
  __shared__ unsigned char smem[2 * STAGE_BYTES];   // 64 KB double buffer

  const int tid  = threadIdx.x;
  const int lane = tid & 31;
  const int wv   = tid >> 5;
  const int r0   = (blockIdx.x * 8 + wv) * 16;
  const int mrow = lane & 15;
  const int hi   = lane >> 4;

  // ---- Load A once: fp32 -> bf16 fragments for all 16 K-chunks ----
  const float* qrow = q + (size_t)(r0 + mrow) * D_MODEL;
  v16bf a[16];
#pragma unroll
  for (int kc = 0; kc < 16; ++kc) {
    const float* p = qrow + kc * 32 + hi * 16;
    float4 x0 = *(const float4*)(p);
    float4 x1 = *(const float4*)(p + 4);
    float4 x2 = *(const float4*)(p + 8);
    float4 x3 = *(const float4*)(p + 12);
    v16bf t;
    t[0]  = f2bf(x0.x); t[1]  = f2bf(x0.y); t[2]  = f2bf(x0.z); t[3]  = f2bf(x0.w);
    t[4]  = f2bf(x1.x); t[5]  = f2bf(x1.y); t[6]  = f2bf(x1.z); t[7]  = f2bf(x1.w);
    t[8]  = f2bf(x2.x); t[9]  = f2bf(x2.y); t[10] = f2bf(x2.z); t[11] = f2bf(x2.w);
    t[12] = f2bf(x3.x); t[13] = f2bf(x3.y); t[14] = f2bf(x3.z); t[15] = f2bf(x3.w);
    a[kc] = t;
  }

  // ---- Async-copy address plan: 4 x 32 B chunks per thread per stage ----
  unsigned gOff[4];   // byte offset inside a stage's global block
  unsigned lOff[4];   // byte offset inside one LDS buffer
#pragma unroll
  for (int j = 0; j < 4; ++j) {
    const int c    = tid + 256 * j;          // chunk id 0..1023
    const int tile = c >> 9;                 // 0 or 1
    const int cc   = c & 511;
    const int kc   = cc >> 5;
    const int l    = cc & 31;
    const int row  = tile * 16 + (l & 15);   // item row within stage
    const int colb = kc * 64 + (l >> 4) * 32;
    gOff[j] = (unsigned)(row * (D_MODEL * 2) + colb);
    lOff[j] = (unsigned)(c * 32);
  }
  const unsigned long long gBase = (unsigned long long)(uintptr_t)itemsBf;
  const unsigned smemBase = (unsigned)(uintptr_t)(void*)smem;

  // issue one stage's async global->LDS copy (8 b128 async ops per thread)
  auto issue_stage = [&](int stage, unsigned bufBase) {
    const unsigned long long g =
        gBase + (unsigned long long)stage * STAGE_BYTES;
#pragma unroll
    for (int j = 0; j < 4; ++j) {
      unsigned long long ga = g + gOff[j];
      unsigned la = bufBase + lOff[j];
      asm volatile("global_load_async_to_lds_b128 %0, %1, off"
                   :: "v"(la), "v"(ga) : "memory");
      asm volatile("global_load_async_to_lds_b128 %0, %1, off offset:16"
                   :: "v"(la), "v"(ga) : "memory");
    }
  };

  float best[8];
  int   bidx[8];
#pragma unroll
  for (int v = 0; v < 8; ++v) { best[v] = -3.4e38f; bidx[v] = 0; }

  issue_stage(0, smemBase);

  for (int s = 0; s < N_STAGES; ++s) {
    // our async writes for buffer (s&1) have landed in LDS
    asm volatile("s_wait_asynccnt 0" ::: "memory");
    // everyone's writes landed; everyone finished reading the other buffer
    __syncthreads();
    if (s + 1 < N_STAGES)
      issue_stage(s + 1, smemBase + (unsigned)(((s + 1) & 1) * STAGE_BYTES));

    const unsigned char* buf = smem + (s & 1) * STAGE_BYTES;
    const unsigned char* f0  = buf + (unsigned)lane * 32;
    const unsigned char* f1  = f0 + TILE_BYTES;

    v8f acc0 = {};
    v8f acc1 = {};
#pragma unroll
    for (int kc = 0; kc < 16; ++kc) {
      v16bf b0 = *(const v16bf*)(f0 + kc * 1024);
      v16bf b1 = *(const v16bf*)(f1 + kc * 1024);
      acc0 = __builtin_amdgcn_wmma_f32_16x16x32_bf16(
          false, a[kc], false, b0, (short)0, acc0, false, false);
      acc1 = __builtin_amdgcn_wmma_f32_16x16x32_bf16(
          false, a[kc], false, b1, (short)0, acc1, false, false);
    }

    const int col0 = s * STAGE_ROWS + mrow;
    const int col1 = col0 + 16;
#pragma unroll
    for (int v = 0; v < 8; ++v) {
      if (acc0[v] > best[v]) { best[v] = acc0[v]; bidx[v] = col0; }
      if (acc1[v] > best[v]) { best[v] = acc1[v]; bidx[v] = col1; }
    }
  }

  // Final 16-lane max/argmax reduction inside each half-wave
#pragma unroll
  for (int v = 0; v < 8; ++v) {
    float bs = best[v];
    int   bi = bidx[v];
#pragma unroll
    for (int off = 8; off >= 1; off >>= 1) {
      float os = __shfl_xor(bs, off, 32);
      int   oi = __shfl_xor(bi, off, 32);
      if (os > bs || (os == bs && oi < bi)) { bs = os; bi = oi; }
    }
    if (mrow == 0) idxOut[r0 + v + hi * 8] = bi;
  }
}

// ---------------------------------------------------------------------------
// Kernel 4: gather + squared-diff, per-row wave reduction, f64 atomic accum
// ---------------------------------------------------------------------------
__global__ void loss_kernel(const float* __restrict__ q,
                            const float* __restrict__ items,
                            const int* __restrict__ idx,
                            double* __restrict__ accum) {
  const int row  = blockIdx.x * 8 + (threadIdx.x >> 5);
  const int lane = threadIdx.x & 31;
  const float* qrow = q + (size_t)row * D_MODEL;
  const float* grow = items + (size_t)idx[row] * D_MODEL;

  float s = 0.0f;
#pragma unroll
  for (int j = 0; j < 4; ++j) {
    float4 aq = *(const float4*)(qrow + lane * 4 + 128 * j);
    float4 ag = *(const float4*)(grow + lane * 4 + 128 * j);
    float dx = aq.x - ag.x, dy = aq.y - ag.y, dz = aq.z - ag.z, dw = aq.w - ag.w;
    s += dx * dx + dy * dy + dz * dz + dw * dw;
  }
#pragma unroll
  for (int off = 16; off >= 1; off >>= 1) s += __shfl_xor(s, off, 32);
  if (lane == 0) atomicAdd(accum, (double)s);      // global_atomic_add_f64
}

// ---------------------------------------------------------------------------
// Kernel 5: finalize mean
// ---------------------------------------------------------------------------
__global__ void finalize_kernel(const double* __restrict__ accum,
                                float* __restrict__ out) {
  out[0] = (float)(*accum * (1.0 / ((double)ROWS_TOT * (double)D_MODEL)));
}

extern "C" void kernel_launch(void* const* d_in, const int* in_sizes, int n_in,
                              void* d_out, int out_size, void* d_ws, size_t ws_size,
                              hipStream_t stream) {
  const float* q     = (const float*)d_in[0];   // [32, 1024, 512] fp32
  const float* items = (const float*)d_in[1];   // [4096, 512] fp32

  char* ws = (char*)d_ws;
  unsigned short* itemsBf = (unsigned short*)ws;                          // 4 MB
  int*    idx   = (int*)   (ws + (size_t)M_ITEMS * D_MODEL * 2);          // 128 KB
  double* accum = (double*)(ws + (size_t)M_ITEMS * D_MODEL * 2
                               + (size_t)ROWS_TOT * 4);                   // 8 B

  cvt_items_kernel<<<(M_ITEMS * D_MODEL) / 4 / 256, 256, 0, stream>>>(items, itemsBf);
  init_kernel<<<1, 1, 0, stream>>>(accum);
  gemm_argmax_kernel<<<ROWS_TOT / 128, 256, 0, stream>>>(q, itemsBf, idx);
  loss_kernel<<<ROWS_TOT / 8, 256, 0, stream>>>(q, items, idx, accum);
  finalize_kernel<<<1, 1, 0, stream>>>(accum, (float*)d_out);
}
